// CausalLTXAttention_44401371906438
// MI455X (gfx1250) — matrix operand order repaired and checked by reading, not compiled, not run on hardware
//
#include <hip/hip_runtime.h>
#include <stdint.h>

// ---------------- problem constants ----------------
#define B_     2
#define L_     2048
#define D_     2048
#define H_     16
#define DH_    64
#define INNER_ 1024
#define WINDOW_ 512
#define SINK_  1
#define EPS_   1e-6f
#define LP_    (L_ + 64)   // padded key dim for transposed V (safe overread)

typedef __attribute__((ext_vector_type(16))) __bf16 v16bf;
typedef __attribute__((ext_vector_type(8)))  __bf16 v8bf;
typedef __attribute__((ext_vector_type(8)))  float  v8f;
typedef __attribute__((ext_vector_type(4)))  float  v4f;

static __device__ __forceinline__ __bf16 f2bf(float f) {
  unsigned u = __builtin_bit_cast(unsigned, f);
  u += 0x7fffu + ((u >> 16) & 1u);          // round-to-nearest-even
  unsigned short h = (unsigned short)(u >> 16);
  return __builtin_bit_cast(__bf16, h);
}

static __device__ __forceinline__ v8f wmma_bf16(v16bf a, v16bf b, v8f c) {
  // D = A(16x32) * B(32x16) + C, fp32 accumulate
  return __builtin_amdgcn_wmma_f32_16x16x32_bf16(false, a, false, b, (short)0, c,
                                                 false, false);
}

static __device__ __forceinline__ v16bf load_a_tile(const __bf16* row, int k0,
                                                    int half) {
  // A operand, contiguous k order chosen here (B matches): row m=lane%16; element i maps to
  // K = k0 + half*16 + i  -> one contiguous 32-byte run.
  v8bf lo = *(const v8bf*)(row + k0 + half * 16);
  v8bf hi = *(const v8bf*)(row + k0 + half * 16 + 8);
  v16bf a;
#pragma unroll
  for (int i = 0; i < 8; ++i) { a[i] = lo[i]; a[i + 8] = hi[i]; }
  return a;
}

// ---------------- fp32 -> bf16 bulk convert ----------------
__global__ __launch_bounds__(256) void k_cvt_bf16(const float* __restrict__ src,
                                                  __bf16* __restrict__ dst, int n4) {
  int i = blockIdx.x * 256 + threadIdx.x;
  if (i < n4) {
    v4f v = *(const v4f*)(src + 4 * (size_t)i);
    dst[4 * (size_t)i + 0] = f2bf(v[0]);
    dst[4 * (size_t)i + 1] = f2bf(v[1]);
    dst[4 * (size_t)i + 2] = f2bf(v[2]);
    dst[4 * (size_t)i + 3] = f2bf(v[3]);
  }
}

// ---------------- GEMM: C[m,n] = sum_k A[m,k]*W[n,k] + bias[n] ----------------
// A: (M,K) bf16 row-major, W: (N,K) bf16 row-major, C: (M,N) fp32.
// Block = 128 thr = 4 waves stacked on M; each wave: 32(M) x 64(N) tile
// (two 16-row A subtiles share the four B operands -> 8 WMMAs / 12 b128 loads).
__global__ __launch_bounds__(128) void k_gemm_bf16(const __bf16* __restrict__ A,
                                                   const __bf16* __restrict__ W,
                                                   const float* __restrict__ bias,
                                                   float* __restrict__ C,
                                                   int K, int ldc) {
  const int lane = threadIdx.x & 31;
  const int wid  = threadIdx.x >> 5;
  const int half = lane >> 4;   // which 16-lane half
  const int l16  = lane & 15;
  const int m0 = blockIdx.x * 128 + wid * 32;
  const int n0 = blockIdx.y * 64;

  const __bf16* arow0 = A + (size_t)(m0 + l16) * K;        // rows m0..m0+15
  const __bf16* arow1 = A + (size_t)(m0 + 16 + l16) * K;   // rows m0+16..m0+31
  const __bf16* wrow  = W + (size_t)(n0 + l16) * K;
  v8f acc[2][4] = {{v8f{}, v8f{}, v8f{}, v8f{}}, {v8f{}, v8f{}, v8f{}, v8f{}}};

  for (int k0 = 0; k0 < K; k0 += 32) {
    // speculative prefetch a few K-steps ahead (global_prefetch_b8)
    __builtin_prefetch(arow0 + k0 + 128, 0, 3);
    __builtin_prefetch(arow1 + k0 + 128, 0, 3);
    __builtin_prefetch(wrow + k0 + 128, 0, 3);

    v16bf a0 = load_a_tile(arow0, k0, half);
    v16bf a1 = load_a_tile(arow1, k0, half);

    // B operand: B[k][n] = W[n][k]; lane holds col n=lane%16, K=half*16+i (contig)
    const __bf16* wbase = wrow + k0 + half * 16;
#pragma unroll
    for (int t = 0; t < 4; ++t) {
      v16bf b = *(const v16bf*)(wbase + (size_t)(16 * t) * K);
      acc[0][t] = wmma_bf16(a0, b, acc[0][t]);
      acc[1][t] = wmma_bf16(a1, b, acc[1][t]);
    }
  }

#pragma unroll
  for (int s = 0; s < 2; ++s) {
#pragma unroll
    for (int t = 0; t < 4; ++t) {
      int n = n0 + 16 * t + l16;
      float bv = bias[n];
#pragma unroll
      for (int r = 0; r < 8; ++r) {      // C layout: row = r + 8*half, col = lane%16
        int row = m0 + 16 * s + r + 8 * half;
        C[(size_t)row * ldc + n] = acc[s][t][r] + bv;
      }
    }
  }
}

// ---------------- QKV epilogue: RMSNorm + RoPE + scale, bf16 pack ----------------
// One block per (b,l) row; 256 thr x 4 elems. Writes q,k (B,L,INNER) bf16 and
// V transposed (B,H,DH,LP) bf16 for contiguous PV B-operand loads.
__global__ __launch_bounds__(256) void k_qkv_epilogue(
    const float* __restrict__ qf, const float* __restrict__ kf,
    const float* __restrict__ vf, const float* __restrict__ cosp,
    const float* __restrict__ sinp, const float* __restrict__ lls,
    const float* __restrict__ qn_w, const float* __restrict__ kn_w,
    __bf16* __restrict__ qb, __bf16* __restrict__ kbb, __bf16* __restrict__ vt) {
  const int row = blockIdx.x;             // b*L + l
  const int b = row / L_, l = row % L_;
  const int t = threadIdx.x;
  const int e0 = t * 4;

  v4f q = *(const v4f*)(qf + (size_t)row * INNER_ + e0);
  v4f k = *(const v4f*)(kf + (size_t)row * INNER_ + e0);

  float sq = q[0]*q[0] + q[1]*q[1] + q[2]*q[2] + q[3]*q[3];
  float sk = k[0]*k[0] + k[1]*k[1] + k[2]*k[2] + k[3]*k[3];
#pragma unroll
  for (int m = 1; m < 32; m <<= 1) {
    sq += __shfl_xor(sq, m, 32);
    sk += __shfl_xor(sk, m, 32);
  }
  __shared__ float red[2][8];
  int wid = t >> 5, lid = t & 31;
  if (lid == 0) { red[0][wid] = sq; red[1][wid] = sk; }
  __syncthreads();
  float tq = 0.f, tk = 0.f;
#pragma unroll
  for (int i = 0; i < 8; ++i) { tq += red[0][i]; tk += red[1][i]; }
  const float qs = rsqrtf(tq * (1.f / INNER_) + EPS_);
  const float ks = rsqrtf(tk * (1.f / INNER_) + EPS_);
  const float qscale = lls[l] * 0.125f;   // logit_log_scale * 1/sqrt(DH)

  const float* cr = cosp + (size_t)l * INNER_ + e0;
  const float* sr = sinp + (size_t)l * INNER_ + e0;
  float qn[4], kn[4];
#pragma unroll
  for (int i = 0; i < 4; ++i) {
    qn[i] = q[i] * qs * qn_w[e0 + i];
    kn[i] = k[i] * ks * kn_w[e0 + i];
  }
  float qo[4], ko[4];
  qo[0] = qn[0]*cr[0] - qn[1]*sr[0];  qo[1] = qn[1]*cr[1] + qn[0]*sr[1];
  qo[2] = qn[2]*cr[2] - qn[3]*sr[2];  qo[3] = qn[3]*cr[3] + qn[2]*sr[3];
  ko[0] = kn[0]*cr[0] - kn[1]*sr[0];  ko[1] = kn[1]*cr[1] + kn[0]*sr[1];
  ko[2] = kn[2]*cr[2] - kn[3]*sr[2];  ko[3] = kn[3]*cr[3] + kn[2]*sr[3];

#pragma unroll
  for (int i = 0; i < 4; ++i) {
    int e = e0 + i;
    qb[(size_t)row * INNER_ + e] = f2bf(qo[i] * qscale);
    kbb[(size_t)row * INNER_ + e] = f2bf(ko[i]);
    int h = e >> 6, d = e & 63;
    vt[((size_t)(b * H_ + h) * DH_ + d) * LP_ + l] =
        f2bf(vf[(size_t)row * INNER_ + e]);
  }
}

// ---------------- banded flash attention ----------------
// Grid (L/16, B*H), 1 wave/block. Online softmax over sink block + window blocks.
__global__ __launch_bounds__(32) void k_attn(const __bf16* __restrict__ qb,
                                             const __bf16* __restrict__ kbp,
                                             const __bf16* __restrict__ vt,
                                             __bf16* __restrict__ ob) {
  const int qt = blockIdx.x, bh = blockIdx.y;
  const int b = bh >> 4, h = bh & (H_ - 1);
  const int q0 = qt * 16;
  const int lane = threadIdx.x;
  const int half = lane >> 4;
  const int l16 = lane & 15;

  // Q in A layout (row = lane%16, K over head dim, two K-steps of 32)
  const __bf16* qrow = qb + ((size_t)(b * L_ + q0 + l16) * INNER_) + h * DH_;
  v16bf aq0 = load_a_tile(qrow, 0, half);
  v16bf aq1 = load_a_tile(qrow, 32, half);

  v8f o[4] = {v8f{}, v8f{}, v8f{}, v8f{}};
  float mrow[8], lrow[8];
#pragma unroll
  for (int r = 0; r < 8; ++r) { mrow[r] = -1e30f; lrow[r] = 0.f; }

  __shared__ __bf16 pbuf[16][32];

  const int bend = (q0 + 15) >> 5;
  int first = q0 - WINDOW_ + 1; if (first < 0) first = 0;
  int bstart = first >> 5; if (bstart < 1) bstart = 1;   // block 0 done separately

  for (int bi = 0; bi <= bend; bi = (bi == 0 ? bstart : bi + 1)) {
    const int kb0 = bi * 32;
    v8f s0 = {}, s1 = {};
    {  // keys kb0 .. kb0+15
      int kidx = kb0 + l16; if (kidx > L_ - 1) kidx = L_ - 1;
      const __bf16* krow = kbp + ((size_t)(b * L_ + kidx) * INNER_) + h * DH_;
      v16bf bk0 = *(const v16bf*)(krow + half * 16);
      v16bf bk1 = *(const v16bf*)(krow + 32 + half * 16);
      s0 = wmma_bf16(aq0, bk0, s0);
      s0 = wmma_bf16(aq1, bk1, s0);
    }
    {  // keys kb0+16 .. kb0+31
      int kidx = kb0 + 16 + l16; if (kidx > L_ - 1) kidx = L_ - 1;
      const __bf16* krow = kbp + ((size_t)(b * L_ + kidx) * INNER_) + h * DH_;
      v16bf bk0 = *(const v16bf*)(krow + half * 16);
      v16bf bk1 = *(const v16bf*)(krow + 32 + half * 16);
      s1 = wmma_bf16(aq0, bk0, s1);
      s1 = wmma_bf16(aq1, bk1, s1);
    }

    // mask + online softmax (row m = r + 8*half lives in one 16-lane half)
    float p0[8], p1[8];
#pragma unroll
    for (int r = 0; r < 8; ++r) {
      int qi = q0 + r + 8 * half;
      int k0i = kb0 + l16;
      int k1i = kb0 + 16 + l16;
      bool ok0 = (k0i <= qi) && ((k0i + WINDOW_ > qi) || (k0i < SINK_));
      bool ok1 = (k1i <= qi) && ((k1i + WINDOW_ > qi) || (k1i < SINK_));
      float v0 = ok0 ? s0[r] : -1e30f;
      float v1 = ok1 ? s1[r] : -1e30f;
      float rm = fmaxf(v0, v1);
      rm = fmaxf(rm, __shfl_xor(rm, 1, 32));
      rm = fmaxf(rm, __shfl_xor(rm, 2, 32));
      rm = fmaxf(rm, __shfl_xor(rm, 4, 32));
      rm = fmaxf(rm, __shfl_xor(rm, 8, 32));
      float mnew = fmaxf(mrow[r], rm);
      float alpha = __expf(mrow[r] - mnew);
      float e0v = __expf(v0 - mnew);
      float e1v = __expf(v1 - mnew);
      float rs = e0v + e1v;
      rs += __shfl_xor(rs, 1, 32);
      rs += __shfl_xor(rs, 2, 32);
      rs += __shfl_xor(rs, 4, 32);
      rs += __shfl_xor(rs, 8, 32);
      lrow[r] = lrow[r] * alpha + rs;
      mrow[r] = mnew;
      o[0][r] *= alpha; o[1][r] *= alpha; o[2][r] *= alpha; o[3][r] *= alpha;
      p0[r] = e0v; p1[r] = e1v;
    }

    // C-layout P -> A-layout through LDS
    __syncthreads();
#pragma unroll
    for (int r = 0; r < 8; ++r) {
      int rr = r + 8 * half;
      pbuf[rr][l16] = f2bf(p0[r]);
      pbuf[rr][16 + l16] = f2bf(p1[r]);
    }
    __syncthreads();
    v16bf ap;
    {
      v8bf plo = *(const v8bf*)(&pbuf[l16][half * 16]);
      v8bf phi = *(const v8bf*)(&pbuf[l16][half * 16 + 8]);
#pragma unroll
      for (int i = 0; i < 8; ++i) { ap[i] = plo[i]; ap[i + 8] = phi[i]; }
    }

    // V^T B-operand: lane holds d = 16t + lane%16, K = kb0 + half*16 + i (contig)
    const __bf16* vbase =
        vt + ((size_t)((b * H_ + h) * DH_ + l16) * LP_) + kb0 + half * 16;
#pragma unroll
    for (int t = 0; t < 4; ++t) {
      v16bf bv = *(const v16bf*)(vbase + (size_t)(16 * t) * LP_);
      o[t] = wmma_bf16(ap, bv, o[t]);
    }
  }

  // normalize + store (B,L,INNER) bf16
#pragma unroll
  for (int r = 0; r < 8; ++r) {
    float inv = 1.f / lrow[r];
    int row = q0 + r + 8 * half;
    __bf16* orow = ob + ((size_t)(b * L_ + row) * INNER_) + h * DH_;
    orow[l16]      = f2bf(o[0][r] * inv);
    orow[16 + l16] = f2bf(o[1][r] * inv);
    orow[32 + l16] = f2bf(o[2][r] * inv);
    orow[48 + l16] = f2bf(o[3][r] * inv);
  }
}

// ---------------- host launcher ----------------
extern "C" void kernel_launch(void* const* d_in, const int* in_sizes, int n_in,
                              void* d_out, int out_size, void* d_ws, size_t ws_size,
                              hipStream_t stream) {
  (void)in_sizes; (void)n_in; (void)out_size;
  const float* x    = (const float*)d_in[0];
  const float* cosp = (const float*)d_in[1];
  const float* sinp = (const float*)d_in[2];
  const float* lls  = (const float*)d_in[3];
  const float* Wq   = (const float*)d_in[4];
  const float* bq   = (const float*)d_in[5];
  const float* Wk   = (const float*)d_in[6];
  const float* bk   = (const float*)d_in[7];
  const float* Wv   = (const float*)d_in[8];
  const float* bv   = (const float*)d_in[9];
  const float* qn_w = (const float*)d_in[10];
  const float* kn_w = (const float*)d_in[11];
  const float* Wo   = (const float*)d_in[12];
  const float* bo   = (const float*)d_in[13];
  float* out = (float*)d_out;

  char* ws = (char*)d_ws;
  size_t off = 0;
  auto carve = [&](size_t bytes) -> char* {
    char* p = ws + off;
    off += (bytes + 255) & ~(size_t)255;
    return p;
  };
  const size_t nx = (size_t)B_ * L_ * D_;       // 8M
  const size_t nw = (size_t)INNER_ * D_;        // 2M
  const size_t nq = (size_t)B_ * L_ * INNER_;   // 4M
  const size_t nvt = (size_t)B_ * H_ * DH_ * LP_;

  __bf16* xb  = (__bf16*)carve(nx * 2);
  __bf16* wqb = (__bf16*)carve(nw * 2);
  __bf16* wkb = (__bf16*)carve(nw * 2);
  __bf16* wvb = (__bf16*)carve(nw * 2);
  __bf16* wob = (__bf16*)carve(nw * 2);
  float*  qf  = (float*)carve(nq * 4);
  float*  kf  = (float*)carve(nq * 4);
  float*  vf  = (float*)carve(nq * 4);
  __bf16* qbb = (__bf16*)carve(nq * 2);
  __bf16* kbb = (__bf16*)carve(nq * 2);
  __bf16* vt  = (__bf16*)carve(nvt * 2);
  __bf16* ob  = (__bf16*)carve(nq * 2);
  if (off > ws_size) return;  // workspace too small; nothing safe to do

  // 1) bf16 conversions
  k_cvt_bf16<<<(int)(nx / 4 / 256), 256, 0, stream>>>(x, xb, (int)(nx / 4));
  k_cvt_bf16<<<(int)(nw / 4 / 256), 256, 0, stream>>>(Wq, wqb, (int)(nw / 4));
  k_cvt_bf16<<<(int)(nw / 4 / 256), 256, 0, stream>>>(Wk, wkb, (int)(nw / 4));
  k_cvt_bf16<<<(int)(nw / 4 / 256), 256, 0, stream>>>(Wv, wvb, (int)(nw / 4));
  k_cvt_bf16<<<(int)(nw / 4 / 256), 256, 0, stream>>>(Wo, wob, (int)(nw / 4));

  // 2) QKV projections: (B*L,D) x (INNER,D)^T -> (B*L,INNER) fp32
  dim3 gp(B_ * L_ / 128, INNER_ / 64);
  k_gemm_bf16<<<gp, 128, 0, stream>>>(xb, wqb, bq, qf, D_, INNER_);
  k_gemm_bf16<<<gp, 128, 0, stream>>>(xb, wkb, bk, kf, D_, INNER_);
  k_gemm_bf16<<<gp, 128, 0, stream>>>(xb, wvb, bv, vf, D_, INNER_);

  // 3) RMSNorm + RoPE + scale epilogue
  k_qkv_epilogue<<<B_ * L_, 256, 0, stream>>>(qf, kf, vf, cosp, sinp, lls, qn_w,
                                              kn_w, qbb, kbb, vt);

  // 4) banded flash attention
  dim3 ga(L_ / 16, B_ * H_);
  k_attn<<<ga, 32, 0, stream>>>(qbb, kbb, vt, ob);

  // 5) output projection: (B*L,INNER) x (D,INNER)^T + bo -> (B,L,D) fp32
  dim3 go(B_ * L_ / 128, D_ / 64);
  k_gemm_bf16<<<go, 128, 0, stream>>>(ob, wob, bo, out, INNER_, D_);
}
